// Decoder_38740605010061
// MI455X (gfx1250) — matrix operand order repaired and checked by reading, not compiled, not run on hardware
//
#include <hip/hip_runtime.h>
#include <hip/hip_bf16.h>

// ---------------------------------------------------------------------------
// Problem constants (from reference)
// ---------------------------------------------------------------------------
#define TT      1024   // encoder time steps
#define NN      128    // batch
#define VV      35     // vocab
#define HH      512    // lstm1 hidden
#define KS      128    // key size / lstm2 hidden
#define VS      128    // value size
#define MAXLEN  250

#define K1      1152   // lstm1 fused K  (512 emb + 128 vmean + 512 h1)
#define K2      640    // lstm2 fused K  (512 h1 + 128 h2)

typedef __attribute__((ext_vector_type(16))) __bf16 v16bf;
typedef __attribute__((ext_vector_type(8)))  __bf16 v8bf;
typedef __attribute__((ext_vector_type(8)))  float  v8f;

__device__ __forceinline__ float sigmoidf_(float x) { return 1.0f / (1.0f + __expf(-x)); }

// CDNA5 async global->LDS copy (ASYNCcnt-tracked), 16B per lane.
__device__ __forceinline__ void async_load_b128(unsigned int lds_off, const void* gaddr) {
    asm volatile("global_load_async_to_lds_b128 %0, %1, off"
                 :: "v"(lds_off), "v"(gaddr)
                 : "memory");
}
__device__ __forceinline__ void wait_async_le1() {
    asm volatile("s_wait_asynccnt 0x1" ::: "memory");
}
__device__ __forceinline__ void wait_async_le0() {
    asm volatile("s_wait_asynccnt 0x0" ::: "memory");
}

// ---------------------------------------------------------------------------
// One-time init kernels
// ---------------------------------------------------------------------------
__global__ void init_w1_kernel(const float* __restrict__ W_ih1,
                               const float* __restrict__ W_hh1,
                               __bf16* __restrict__ Wcat1) {
    int idx = blockIdx.x * 256 + threadIdx.x;          // (4H)*K1 threads
    int j = idx / K1, k = idx % K1;
    float v = (k < 640) ? W_ih1[j * 640 + k] : W_hh1[j * 512 + (k - 640)];
    Wcat1[idx] = (__bf16)v;
}

__global__ void init_w2_kernel(const float* __restrict__ W_ih2,
                               const float* __restrict__ W_hh2,
                               __bf16* __restrict__ Wcat2) {
    int idx = blockIdx.x * 256 + threadIdx.x;          // (4KS)*K2 threads
    int j = idx / K2, k = idx % K2;
    float v = (k < 512) ? W_ih2[j * 512 + k] : W_hh2[j * 128 + (k - 512)];
    Wcat2[idx] = (__bf16)v;
}

// key_bf16[n][t][k] = enc_key[t][n][k]
__global__ void init_key_kernel(const float* __restrict__ ek, __bf16* __restrict__ key) {
    int idx = blockIdx.x * 256 + threadIdx.x;          // N*T*KS threads
    int k = idx & (KS - 1);
    int t = (idx >> 7) & (TT - 1);
    int n = idx >> 17;
    key[idx] = (__bf16)ek[((size_t)t * NN + n) * KS + k];
}

// valT_bf16[n][v][t] = enc_value[t][n][v]
__global__ void init_valT_kernel(const float* __restrict__ ev, __bf16* __restrict__ valT) {
    int idx = blockIdx.x * 256 + threadIdx.x;          // N*VS*T threads
    int t = idx & (TT - 1);
    int v = (idx >> 10) & (VS - 1);
    int n = idx >> 17;
    valT[idx] = (__bf16)ev[((size_t)t * NN + n) * VS + v];
}

// values_mean -> bf16 slice of both xcat1 ping-pong buffers
__global__ void init_vmean_kernel(const float* __restrict__ ev,
                                  __bf16* __restrict__ x1a, __bf16* __restrict__ x1b) {
    int idx = blockIdx.x * 256 + threadIdx.x;          // N*VS threads
    int n = idx >> 7, v = idx & (VS - 1);
    float s = 0.f;
    for (int t = 0; t < TT; ++t) s += ev[((size_t)t * NN + n) * VS + v];
    __bf16 m = (__bf16)(s * (1.0f / (float)TT));
    x1a[(size_t)n * K1 + 512 + v] = m;
    x1b[(size_t)n * K1 + 512 + v] = m;
}

__global__ void init_zero_kernel(float* __restrict__ c1, float* __restrict__ c2,
                                 float* __restrict__ h2,
                                 __bf16* __restrict__ x1a, __bf16* __restrict__ x1b,
                                 __bf16* __restrict__ x2a, __bf16* __restrict__ x2b) {
    int idx = blockIdx.x * 256 + threadIdx.x;          // grid covers N*K1 = 147456
    if (idx < NN * HH) c1[idx] = 0.f;
    if (idx < NN * KS) { c2[idx] = 0.f; h2[idx] = 0.f; }
    if (idx < NN * K1) { x1a[idx] = (__bf16)0.f; x1b[idx] = (__bf16)0.f; }
    if (idx < NN * K2) { x2a[idx] = (__bf16)0.f; x2b[idx] = (__bf16)0.f; }
}

// ---------------------------------------------------------------------------
// Fused bf16-WMMA GEMM + LSTM cell.
//   gates(128 x 4*HDIM) = A(128 x K) * W(4*HDIM x K)^T ; then cell pointwise.
// Block = 256 threads = 8 waves. Block bx owns j-columns [bx*16, bx*16+16) of
// EACH of the 4 gate sections (i,f,g,o), so a wave's 4 accumulators are the
// 4 gates of one 16x16 slab -> full cell update in registers.
// B slice (64 rows x 32 K, 4KB) staged in LDS via async copy, double-buffered.
// ---------------------------------------------------------------------------
template <int K, int HDIM, bool L1>
__global__ __launch_bounds__(256) void gemm_cell_kernel(
    const __bf16* __restrict__ A,       // 128 x K        (xcat cur)
    const __bf16* __restrict__ W,       // (4*HDIM) x K   (bf16 fused weights)
    const float* __restrict__ b_ih, const float* __restrict__ b_hh,
    float* __restrict__ cstate,         // 128 x HDIM
    __bf16* __restrict__ hdst0,         // L1: xcat1_nxt (h1 slot) | L2: xcat2_nxt (h2 slot)
    __bf16* __restrict__ hdst1,         // L1: xcat2_cur (h1 slot) | L2: unused
    float*  __restrict__ hf32) {        // L2: h2 fp32 for attention | L1: unused
    __shared__ __attribute__((aligned(32))) __bf16 bs[2 * 64 * 32];   // 8 KB

    const int tid  = threadIdx.x;
    const int lane = tid & 31;
    const int mb   = tid >> 5;                 // batch tile 0..7
    const int J0   = blockIdx.x * 16;          // j-column slab

    const int mrow = mb * 16 + (lane & 15);    // A row
    const int kbA  = (lane >> 4) * 16;         // A frag: lanes 16-31 take K+16
    const int kbB  = (lane >> 4) * 16;         // B frag: lanes 16-31 take K+16
    const int ncol = lane & 15;

    // async staging assignment: thread -> (row rr of B slice, 16B chunk cc)
    const int rr   = tid >> 2;                 // 0..63
    const int cc   = tid & 3;
    const int sect = rr >> 4;                  // gate section 0..3
    const int grow = sect * HDIM + J0 + (rr & 15);
    const __bf16* gsrc = W + (size_t)grow * K + cc * 8;
    const unsigned int lds0 =
        (unsigned int)(uintptr_t)(&bs[0]) + (unsigned int)((rr * 32 + cc * 8) * 2);

    // prologue: fill buffer 0 (k-step 0)
    async_load_b128(lds0, gsrc);

    const __bf16* arow = A + (size_t)mrow * K;
    v8bf alo = *(const v8bf*)(arow + kbA);
    v8bf ahi = *(const v8bf*)(arow + kbA + 8);

    v8f acc_i = {}, acc_f = {}, acc_g = {}, acc_o = {};

    constexpr int NK = K / 32;
    int cur = 0;
    for (int ksi = 0; ksi < NK; ++ksi) {
        if (ksi + 1 < NK) {
            async_load_b128(lds0 + (unsigned int)((cur ^ 1) * 4096), gsrc + (ksi + 1) * 32);
            wait_async_le1();                  // current buffer's fill complete
        } else {
            wait_async_le0();
        }
        __syncthreads();                       // all waves' fills visible

        const __bf16* bb = bs + cur * (64 * 32);
        v16bf b0 = *(const v16bf*)(bb + (0 * 16 + ncol) * 32 + kbB);
        v16bf b1 = *(const v16bf*)(bb + (1 * 16 + ncol) * 32 + kbB);
        v16bf b2 = *(const v16bf*)(bb + (2 * 16 + ncol) * 32 + kbB);
        v16bf b3 = *(const v16bf*)(bb + (3 * 16 + ncol) * 32 + kbB);

        v16bf a = __builtin_shufflevector(alo, ahi, 0, 1, 2, 3, 4, 5, 6, 7,
                                          8, 9, 10, 11, 12, 13, 14, 15);
        if (ksi + 1 < NK) {                    // software-pipeline next A frag
            alo = *(const v8bf*)(arow + (ksi + 1) * 32 + kbA);
            ahi = *(const v8bf*)(arow + (ksi + 1) * 32 + kbA + 8);
        }

        acc_i = __builtin_amdgcn_wmma_f32_16x16x32_bf16(false, a, false, b0, (short)0, acc_i, false, false);
        acc_f = __builtin_amdgcn_wmma_f32_16x16x32_bf16(false, a, false, b1, (short)0, acc_f, false, false);
        acc_g = __builtin_amdgcn_wmma_f32_16x16x32_bf16(false, a, false, b2, (short)0, acc_g, false, false);
        acc_o = __builtin_amdgcn_wmma_f32_16x16x32_bf16(false, a, false, b3, (short)0, acc_o, false, false);

        __syncthreads();                       // done reading buf before refill
        cur ^= 1;
    }

    // ---- fused LSTM cell epilogue (gate order i,f,g,o) ----
    const int j = J0 + ncol;
    const float bi = b_ih[j]            + b_hh[j];
    const float bf = b_ih[HDIM + j]     + b_hh[HDIM + j];
    const float bg = b_ih[2 * HDIM + j] + b_hh[2 * HDIM + j];
    const float bo = b_ih[3 * HDIM + j] + b_hh[3 * HDIM + j];
    const int rbase = (lane >> 4) * 8;
#pragma unroll
    for (int r = 0; r < 8; ++r) {
        const int m = mb * 16 + rbase + r;
        const float gi = acc_i[r] + bi;
        const float gf = acc_f[r] + bf;
        const float gg = acc_g[r] + bg;
        const float go = acc_o[r] + bo;
        const float c  = sigmoidf_(gf) * cstate[m * HDIM + j] + sigmoidf_(gi) * tanhf(gg);
        const float h  = sigmoidf_(go) * tanhf(c);
        cstate[m * HDIM + j] = c;
        if constexpr (L1) {
            hdst0[(size_t)m * K1 + 640 + j] = (__bf16)h;   // h1 slot, next-step xcat1
            hdst1[(size_t)m * K2 + j]       = (__bf16)h;   // h1 slot, this-step xcat2
        } else {
            hdst0[(size_t)m * K2 + 512 + j] = (__bf16)h;   // h2 slot, next-step xcat2
            hf32[m * KS + j] = h;                          // fp32 h2 for attention
        }
    }
}

// ---------------------------------------------------------------------------
// Fused attention (energy -> masked softmax -> context) + output projection
// + greedy argmax + embedding write for the next step. One block per batch n.
// ---------------------------------------------------------------------------
__global__ __launch_bounds__(256) void attn_pred_kernel(
    const __bf16* __restrict__ key,    // (N,T,KS)
    const __bf16* __restrict__ valT,   // (N,VS,T)
    const float* __restrict__ h2,      // (N,KS)
    const int* __restrict__ lens,
    const float* __restrict__ Wout,    // (V, KS+VS)
    const float* __restrict__ bout,
    const float* __restrict__ emb,     // (V, H)
    float* __restrict__ out,           // (N, MAXLEN, V)
    __bf16* __restrict__ xcat1_nxt,    // embedding slot target
    int step) {
    __shared__ float h2s[KS];
    __shared__ float attn[TT];
    __shared__ float red[256];
    __shared__ float ctxs[VS];
    __shared__ int   stok;
    const int n = blockIdx.x;
    const int tid = threadIdx.x;
    if (tid < KS) h2s[tid] = h2[n * KS + tid];
    __syncthreads();
    const int len = lens[n];

    // ---- energy ----
    float lmax = -3.0e38f;
#pragma unroll
    for (int tt = 0; tt < 4; ++tt) {
        int t = tt * 256 + tid;
        float e = -1.0e30f;
        if (t < len) {
            const __bf16* kp = key + ((size_t)n * TT + t) * KS;
            float acc = 0.f;
            for (int k = 0; k < KS; k += 8) {
                v8bf kv = *(const v8bf*)(kp + k);
#pragma unroll
                for (int i = 0; i < 8; ++i) acc += (float)kv[i] * h2s[k + i];
            }
            e = acc;
        }
        attn[t] = e;
        lmax = fmaxf(lmax, e);
    }
    red[tid] = lmax;
    __syncthreads();
    for (int s = 128; s > 0; s >>= 1) {
        if (tid < s) red[tid] = fmaxf(red[tid], red[tid + s]);
        __syncthreads();
    }
    const float m = red[0];
    __syncthreads();

    // ---- softmax ----
    float lsum = 0.f;
#pragma unroll
    for (int tt = 0; tt < 4; ++tt) {
        int t = tt * 256 + tid;
        float w = __expf(attn[t] - m);
        attn[t] = w;
        lsum += w;
    }
    red[tid] = lsum;
    __syncthreads();
    for (int s = 128; s > 0; s >>= 1) {
        if (tid < s) red[tid] += red[tid + s];
        __syncthreads();
    }
    const float scale = 1.0f / red[0];
    __syncthreads();

    // ---- context: 2 threads per value dim, each reduces half of T ----
    const int v = tid & 127, half = tid >> 7;
    const __bf16* vp = valT + ((size_t)n * VS + v) * TT + half * 512;
    const float* ap = attn + half * 512;
    float acc = 0.f;
    for (int t = 0; t < 512; t += 8) {
        v8bf vv = *(const v8bf*)(vp + t);
#pragma unroll
        for (int i = 0; i < 8; ++i) acc += (float)vv[i] * ap[t + i];
    }
    red[tid] = acc;
    __syncthreads();
    if (half == 0) ctxs[v] = (red[v] + red[v + 128]) * scale;
    __syncthreads();

    // ---- output projection ----
    float p = -3.0e38f;
    if (tid < VV) {
        const float* w = Wout + tid * (KS + VS);
        float accp = bout[tid];
        for (int k = 0; k < KS; ++k) accp += h2s[k] * w[k];
        for (int k = 0; k < VS; ++k) accp += ctxs[k] * w[KS + k];
        out[((size_t)n * MAXLEN + step) * VV + tid] = accp;
        p = accp;
    }
    if (tid < 64) red[tid] = p;
    __syncthreads();
    if (tid == 0) {
        int best = 0;
        float bv = red[0];
        for (int k = 1; k < VV; ++k)
            if (red[k] > bv) { bv = red[k]; best = k; }   // first-max tie-break
        stok = best;
    }
    __syncthreads();

    // ---- embedding for next step: 256 threads x 2 elems ----
    const float* erow = emb + (size_t)stok * HH;
    __bf16* dst = xcat1_nxt + (size_t)n * K1;
    const int jj = tid * 2;
    float2 ev = *(const float2*)(erow + jj);
    dst[jj]     = (__bf16)ev.x;
    dst[jj + 1] = (__bf16)ev.y;
}

// ---------------------------------------------------------------------------
// Host driver (all launches on `stream`; graph-capture safe)
// ---------------------------------------------------------------------------
extern "C" void kernel_launch(void* const* d_in, const int* in_sizes, int n_in,
                              void* d_out, int out_size, void* d_ws, size_t ws_size,
                              hipStream_t stream) {
    const float* enc_key   = (const float*)d_in[0];
    const float* enc_value = (const float*)d_in[1];
    const int*   lens      = (const int*)d_in[2];
    const float* emb       = (const float*)d_in[3];
    const float* W_ih1     = (const float*)d_in[4];
    const float* W_hh1     = (const float*)d_in[5];
    const float* b_ih1     = (const float*)d_in[6];
    const float* b_hh1     = (const float*)d_in[7];
    const float* W_ih2     = (const float*)d_in[8];
    const float* W_hh2     = (const float*)d_in[9];
    const float* b_ih2     = (const float*)d_in[10];
    const float* b_hh2     = (const float*)d_in[11];
    const float* W_out     = (const float*)d_in[12];
    const float* b_out     = (const float*)d_in[13];
    float* out = (float*)d_out;

    // workspace carve-up (256B-aligned)
    char* ws = (char*)d_ws;
    size_t off = 0;
    auto carve = [&](size_t bytes) { void* p = ws + off; off += (bytes + 255) & ~(size_t)255; return p; };
    __bf16* key_bf  = (__bf16*)carve((size_t)NN * TT * KS * 2);   // 32 MB
    __bf16* valT_bf = (__bf16*)carve((size_t)NN * VS * TT * 2);   // 32 MB
    __bf16* Wcat1   = (__bf16*)carve((size_t)4 * HH * K1 * 2);
    __bf16* Wcat2   = (__bf16*)carve((size_t)4 * KS * K2 * 2);
    __bf16* xcat1a  = (__bf16*)carve((size_t)NN * K1 * 2);
    __bf16* xcat1b  = (__bf16*)carve((size_t)NN * K1 * 2);
    __bf16* xcat2a  = (__bf16*)carve((size_t)NN * K2 * 2);
    __bf16* xcat2b  = (__bf16*)carve((size_t)NN * K2 * 2);
    float*  c1      = (float*)carve((size_t)NN * HH * 4);
    float*  c2      = (float*)carve((size_t)NN * KS * 4);
    float*  h2      = (float*)carve((size_t)NN * KS * 4);
    (void)ws_size; (void)n_in; (void)in_sizes; (void)out_size;

    // ---- one-time setup ----
    init_key_kernel <<<(NN * TT * KS) / 256, 256, 0, stream>>>(enc_key, key_bf);
    init_valT_kernel<<<(NN * VS * TT) / 256, 256, 0, stream>>>(enc_value, valT_bf);
    init_w1_kernel  <<<(4 * HH * K1) / 256, 256, 0, stream>>>(W_ih1, W_hh1, Wcat1);
    init_w2_kernel  <<<(4 * KS * K2) / 256, 256, 0, stream>>>(W_ih2, W_hh2, Wcat2);
    init_zero_kernel<<<(NN * K1) / 256, 256, 0, stream>>>(c1, c2, h2, xcat1a, xcat1b, xcat2a, xcat2b);
    init_vmean_kernel<<<(NN * VS) / 256, 256, 0, stream>>>(enc_value, xcat1a, xcat1b);

    // ---- 250 sequential decode steps, 3 kernels each ----
    for (int step = 0; step < MAXLEN; ++step) {
        __bf16* x1c = (step & 1) ? xcat1b : xcat1a;
        __bf16* x1n = (step & 1) ? xcat1a : xcat1b;
        __bf16* x2c = (step & 1) ? xcat2b : xcat2a;
        __bf16* x2n = (step & 1) ? xcat2a : xcat2b;

        gemm_cell_kernel<K1, HH, true><<<HH / 16, 256, 0, stream>>>(
            x1c, Wcat1, b_ih1, b_hh1, c1, x1n, x2c, (float*)nullptr);
        gemm_cell_kernel<K2, KS, false><<<KS / 16, 256, 0, stream>>>(
            x2c, Wcat2, b_ih2, b_hh2, c2, x2n, (__bf16*)nullptr, h2);
        attn_pred_kernel<<<NN, 256, 0, stream>>>(
            key_bf, valT_bf, h2, lens, W_out, b_out, emb, out, x1n, step);
    }
}
